// WolfPQVAEMinDistEncoder_61297773248788
// MI455X (gfx1250) — compile-verified, matrix operand order checked by reading, not run on hardware
//
#include <hip/hip_runtime.h>
#include <hip/hip_bf16.h>
#include <math.h>

// ---------------------------------------------------------------------------
// WolfPQVAEMinDistEncoder for MI455X (gfx1250, wave32, WMMA bf16).
//
// All five big GEMMs run on v_wmma_f32_16x16x32_bf16 with fp32 accumulation.
// Weights are converted fp32 -> bf16 once per launch into the workspace.
// Workspace budget (bump allocator, 256B aligned): ~775 MB.
// ---------------------------------------------------------------------------

#define BATCH 2048
#define DIMX  768
#define PQ_M  48
#define PQ_K  256
#define PQ_D  16          // DIMX / PQ_M
#define MIDX  9216
#define HX    6144
#define MKX   12288
#define LN_EPS 1e-5f
#define MIN_DIST_FACTOR 10.0f

typedef __attribute__((ext_vector_type(16))) __bf16 v16bf;
typedef __attribute__((ext_vector_type(8)))  __bf16 v8bf;
typedef __attribute__((ext_vector_type(4)))  __bf16 v4bf;
typedef __attribute__((ext_vector_type(8)))  float  v8f;

union FragBF {
    v16bf v;
    v8bf  h[2];
};

// ---------------------------------------------------------------------------
// fp32 -> bf16 cast, 4 elements per thread (all sizes here divisible by 4)
// ---------------------------------------------------------------------------
__global__ __launch_bounds__(256)
void cast_f32_to_bf16_x4(const float4* __restrict__ in, v4bf* __restrict__ out,
                         int n4) {
    int i = blockIdx.x * 256 + threadIdx.x;
    if (i < n4) {
        float4 f = in[i];
        v4bf b;
        b.x = (__bf16)f.x; b.y = (__bf16)f.y;
        b.z = (__bf16)f.z; b.w = (__bf16)f.w;
        out[i] = b;
    }
}

// ---------------------------------------------------------------------------
// C[M,N] = act(A[M,K] @ W[N,K]^T + bias), bf16 inputs, f32 accumulate.
// Block: 256 threads = 8 waves, arranged 4(M) x 2(N).
// Wave tile: 32(M) x 64(N) -> 2x4 accumulators, 8 WMMAs per 32-wide K step.
// Block tile: 128 x 128. All problem dims are multiples of 128 / 32.
// ACT: 0 = identity, 1 = relu.  OUT_T: float or __bf16.
// ---------------------------------------------------------------------------
template <int ACT, typename OUT_T>
__global__ __launch_bounds__(256)
void gemm_bf16_wmma(const __bf16* __restrict__ A,   // [Mrows, Kd]
                    const __bf16* __restrict__ W,   // [N, Kd]
                    const float*  __restrict__ bias,// [N] or nullptr
                    OUT_T* __restrict__ C,          // [Mrows, N]
                    int N, int Kd) {
    const int lane  = threadIdx.x & 31;
    const int wave  = threadIdx.x >> 5;
    const int lrow  = lane & 15;      // row (A) / col (B) index within tile
    const int khalf = lane >> 4;      // selects K sub-block per ISA layout

    const int wm = wave & 3;          // 0..3
    const int wn = wave >> 2;         // 0..1
    const int m0 = blockIdx.y * 128 + wm * 32;
    const int n0 = blockIdx.x * 128 + wn * 64;

    v8f acc[2][4];
#pragma unroll
    for (int i = 0; i < 2; ++i)
#pragma unroll
        for (int j = 0; j < 4; ++j)
            acc[i][j] = (v8f){0.f, 0.f, 0.f, 0.f, 0.f, 0.f, 0.f, 0.f};

    const __bf16* aBase = A + (size_t)(m0 + lrow) * Kd + khalf * 8;
    const __bf16* wBase = W + (size_t)(n0 + lrow) * Kd + khalf * 8;

    for (int k0 = 0; k0 < Kd; k0 += 32) {
        // A fragments: two 16x32 tiles (rows m0.., m0+16..)
        FragBF af[2];
#pragma unroll
        for (int i = 0; i < 2; ++i) {
            const __bf16* p = aBase + (size_t)i * 16 * Kd + k0;
            af[i].h[0] = *(const v8bf*)(p);
            af[i].h[1] = *(const v8bf*)(p + 16);
        }
        // B fragments: four 32x16 tiles (cols n0.., +16, +32, +48);
        // element (k, n) = W[n][k]: per-lane data is contiguous in W's row.
        FragBF bf[4];
#pragma unroll
        for (int j = 0; j < 4; ++j) {
            const __bf16* p = wBase + (size_t)j * 16 * Kd + k0;
            bf[j].h[0] = *(const v8bf*)(p);
            bf[j].h[1] = *(const v8bf*)(p + 16);
        }
        if (k0 + 32 < Kd) {
            __builtin_prefetch(aBase + k0 + 32, 0, 3);       // global_prefetch_b8
            __builtin_prefetch(wBase + k0 + 32, 0, 3);
        }
#pragma unroll
        for (int i = 0; i < 2; ++i)
#pragma unroll
            for (int j = 0; j < 4; ++j)
                acc[i][j] = __builtin_amdgcn_wmma_f32_16x16x32_bf16(
                    /*neg_a=*/false, af[i].v, /*neg_b=*/false, bf[j].v,
                    /*c_mod=*/(short)0, acc[i][j],
                    /*reuse_a=*/false, /*reuse_b=*/false);
    }

    // Epilogue. C/D layout: vgpr r -> row = r + 8*khalf, col = lane & 15.
#pragma unroll
    for (int i = 0; i < 2; ++i) {
#pragma unroll
        for (int j = 0; j < 4; ++j) {
            const int ncol = n0 + j * 16 + lrow;
            const float bv = bias ? bias[ncol] : 0.f;
#pragma unroll
            for (int r = 0; r < 8; ++r) {
                const int mrow = m0 + i * 16 + r + 8 * khalf;
                float v = acc[i][j][r] + bv;
                if (ACT == 1) v = v > 0.f ? v : 0.f;
                C[(size_t)mrow * N + ncol] = (OUT_T)v;
            }
        }
    }
}

// ---------------------------------------------------------------------------
// LayerNorm over DIMX=768 columns, one block (256 thr) per row; bf16 out.
// ---------------------------------------------------------------------------
__global__ __launch_bounds__(256)
void layernorm_to_bf16(const float* __restrict__ x, const float* __restrict__ g,
                       const float* __restrict__ bta, __bf16* __restrict__ out) {
    const int row = blockIdx.x;
    const float* xr = x + (size_t)row * DIMX;
    __shared__ float sred[16];

    float s = 0.f, sq = 0.f;
    for (int c = threadIdx.x; c < DIMX; c += 256) {
        float v = xr[c];
        s += v;
        sq += v * v;
    }
    for (int o = 16; o; o >>= 1) {
        s  += __shfl_down(s, o, 32);
        sq += __shfl_down(sq, o, 32);
    }
    const int w = threadIdx.x >> 5, l = threadIdx.x & 31;
    if (l == 0) { sred[w] = s; sred[8 + w] = sq; }
    __syncthreads();
    if (threadIdx.x == 0) {
        float ts = 0.f, tq = 0.f;
        for (int i = 0; i < 8; ++i) { ts += sred[i]; tq += sred[8 + i]; }
        float mean = ts * (1.f / DIMX);
        float var  = tq * (1.f / DIMX) - mean * mean;
        sred[0] = mean;
        sred[1] = rsqrtf(var + LN_EPS);
    }
    __syncthreads();
    const float mean = sred[0], rstd = sred[1];
    for (int c = threadIdx.x; c < DIMX; c += 256)
        out[(size_t)row * DIMX + c] =
            (__bf16)((xr[c] - mean) * rstd * g[c] + bta[c]);
}

// ---------------------------------------------------------------------------
// h = tanh(mu + exp(sig) * noise)  -> bf16
// ---------------------------------------------------------------------------
__global__ __launch_bounds__(256)
void reparam_tanh_bf16(const float* __restrict__ mu, const float* __restrict__ sg,
                       const float* __restrict__ noise, __bf16* __restrict__ out,
                       int n) {
    int i = blockIdx.x * 256 + threadIdx.x;
    if (i < n) out[i] = (__bf16)tanhf(fmaf(expf(sg[i]), noise[i], mu[i]));
}

// ---------------------------------------------------------------------------
// PQ argmin: one wave per (b, m); 8 codes per lane; first-index tie-break.
// ---------------------------------------------------------------------------
__global__ __launch_bounds__(256)
void pq_argmin(const float* __restrict__ xrot, const float* __restrict__ codebook,
               int* __restrict__ amin) {
    const int gw = blockIdx.x * 8 + (threadIdx.x >> 5);  // b * PQ_M + m
    if (gw >= BATCH * PQ_M) return;
    const int lane = threadIdx.x & 31;
    const int b = gw / PQ_M, m = gw % PQ_M;

    float xv[PQ_D];
    const float* xs = xrot + (size_t)b * DIMX + m * PQ_D;
#pragma unroll
    for (int i = 0; i < PQ_D; ++i) xv[i] = xs[i];

    const float* cb = codebook + (size_t)m * PQ_K * PQ_D;
    float best = 3.4e38f;
    int   bi   = 0;
    for (int kk = lane; kk < PQ_K; kk += 32) {
        const float* c = cb + (size_t)kk * PQ_D;
        float d = 0.f;
#pragma unroll
        for (int i = 0; i < PQ_D; ++i) {
            float t = c[i] - xv[i];
            d = fmaf(t, t, d);
        }
        if (d < best) { best = d; bi = kk; }   // kk increases -> keeps first
    }
    for (int o = 16; o; o >>= 1) {
        float ob = __shfl_xor(best, o, 32);
        int   oi = __shfl_xor(bi, o, 32);
        if (ob < best || (ob == best && oi < bi)) { best = ob; bi = oi; }
    }
    if (lane == 0) amin[gw] = bi;
}

// ---------------------------------------------------------------------------
// Hard gumbel-softmax output: one wave per (b, m) row of K=256.
// s = (y_hard - y_soft) + y_soft, matching the reference's value path.
// ---------------------------------------------------------------------------
__global__ __launch_bounds__(256)
void gumbel_hard_out(const float* __restrict__ a, const float* __restrict__ gum,
                     const int* __restrict__ amin,
                     const float* __restrict__ interp_param,
                     float* __restrict__ out) {
    const int gw = blockIdx.x * 8 + (threadIdx.x >> 5);  // b * PQ_M + m
    if (gw >= BATCH * PQ_M) return;
    const int lane = threadIdx.x & 31;

    const float interp = 1.f / (1.f + expf(-interp_param[0]));
    const int mi = amin[gw];
    const float* ar = a   + (size_t)gw * PQ_K;
    const float* gr = gum + (size_t)gw * PQ_K;

    float lg[8];
    float lmax = -3.4e38f;
#pragma unroll
    for (int j = 0; j < 8; ++j) {
        const int kk = lane + j * 32;
        float l = (1.f - interp) * ar[kk] +
                  (kk == mi ? interp * MIN_DIST_FACTOR : 0.f) + gr[kk];
        lg[j] = l;
        lmax = fmaxf(lmax, l);
    }
    for (int o = 16; o; o >>= 1) lmax = fmaxf(lmax, __shfl_xor(lmax, o, 32));

    float e[8];
    float sum = 0.f;
#pragma unroll
    for (int j = 0; j < 8; ++j) { e[j] = expf(lg[j] - lmax); sum += e[j]; }
    for (int o = 16; o; o >>= 1) sum += __shfl_xor(sum, o, 32);
    const float inv = 1.f / sum;

    // argmax with first-index tie-break (jnp.argmax semantics)
    float bv = -3.4e38f;
    int   bk = 1 << 30;
#pragma unroll
    for (int j = 0; j < 8; ++j) {
        const int kk = lane + j * 32;
        if (lg[j] > bv) { bv = lg[j]; bk = kk; }
    }
    for (int o = 16; o; o >>= 1) {
        float ov = __shfl_xor(bv, o, 32);
        int   ok = __shfl_xor(bk, o, 32);
        if (ov > bv || (ov == bv && ok < bk)) { bv = ov; bk = ok; }
    }

#pragma unroll
    for (int j = 0; j < 8; ++j) {
        const int kk = lane + j * 32;
        const float ys = e[j] * inv;
        const float yh = (kk == bk) ? 1.f : 0.f;
        out[(size_t)gw * PQ_K + kk] = (yh - ys) + ys;
    }
}

// ---------------------------------------------------------------------------
// Host launcher
// ---------------------------------------------------------------------------
extern "C" void kernel_launch(void* const* d_in, const int* in_sizes, int n_in,
                              void* d_out, int out_size, void* d_ws,
                              size_t ws_size, hipStream_t stream) {
    (void)in_sizes; (void)n_in; (void)out_size; (void)ws_size;

    const float* x       = (const float*)d_in[0];
    const float* codebook= (const float*)d_in[1];
    const float* rot1    = (const float*)d_in[2];
    const float* ln_g    = (const float*)d_in[3];
    const float* ln_b    = (const float*)d_in[4];
    const float* l1_w    = (const float*)d_in[5];
    const float* l1_b    = (const float*)d_in[6];
    const float* mu_w    = (const float*)d_in[7];
    const float* mu_b    = (const float*)d_in[8];
    const float* sig_w   = (const float*)d_in[9];
    const float* sig_b   = (const float*)d_in[10];
    const float* rot2    = (const float*)d_in[11];
    const float* l2_w    = (const float*)d_in[12];
    const float* l2_b    = (const float*)d_in[13];
    const float* interp  = (const float*)d_in[14];
    const float* noise   = (const float*)d_in[15];
    const float* gumbel  = (const float*)d_in[16];
    float* out           = (float*)d_out;

    // Workspace bump allocator (256B aligned); total ~775 MB.
    size_t off = 0;
    auto alloc = [&](size_t bytes) -> void* {
        void* p = (char*)d_ws + off;
        off += (bytes + 255) & ~(size_t)255;
        return p;
    };
    __bf16* x_bf    = (__bf16*)alloc((size_t)BATCH * DIMX * 2);
    __bf16* rot1_bf = (__bf16*)alloc((size_t)DIMX * DIMX * 2);
    __bf16* l1w_bf  = (__bf16*)alloc((size_t)MIDX * DIMX * 2);
    __bf16* muw_bf  = (__bf16*)alloc((size_t)HX * MIDX * 2);
    __bf16* sigw_bf = (__bf16*)alloc((size_t)HX * MIDX * 2);
    __bf16* rot2_bf = (__bf16*)alloc((size_t)HX * HX * 2);
    __bf16* l2w_bf  = (__bf16*)alloc((size_t)MKX * HX * 2);
    float*  x_rot   = (float*)alloc((size_t)BATCH * DIMX * 4);
    __bf16* xn_bf   = (__bf16*)alloc((size_t)BATCH * DIMX * 2);
    __bf16* h1_bf   = (__bf16*)alloc((size_t)BATCH * MIDX * 2);
    float*  mu_pre  = (float*)alloc((size_t)BATCH * HX * 4);
    float*  sig_pre = (float*)alloc((size_t)BATCH * HX * 4);
    __bf16* h_bf    = (__bf16*)alloc((size_t)BATCH * HX * 2);
    __bf16* hrot_bf = (__bf16*)alloc((size_t)BATCH * HX * 2);
    float*  a_act   = (float*)alloc((size_t)BATCH * MKX * 4);
    int*    amin    = (int*)alloc((size_t)BATCH * PQ_M * 4);

    auto cast_bf16 = [&](const float* src, __bf16* dst, size_t n) {
        int n4 = (int)(n / 4);
        cast_f32_to_bf16_x4<<<(n4 + 255) / 256, 256, 0, stream>>>(
            (const float4*)src, (v4bf*)dst, n4);
    };
    cast_bf16(x,     x_bf,    (size_t)BATCH * DIMX);
    cast_bf16(rot1,  rot1_bf, (size_t)DIMX * DIMX);
    cast_bf16(l1_w,  l1w_bf,  (size_t)MIDX * DIMX);
    cast_bf16(mu_w,  muw_bf,  (size_t)HX * MIDX);
    cast_bf16(sig_w, sigw_bf, (size_t)HX * MIDX);
    cast_bf16(rot2,  rot2_bf, (size_t)HX * HX);
    cast_bf16(l2_w,  l2w_bf,  (size_t)MKX * HX);

    const dim3 blk(256);
    // 1) x_rot = x @ rot1^T                       [2048, 768]
    gemm_bf16_wmma<0, float><<<dim3(DIMX / 128, BATCH / 128), blk, 0, stream>>>(
        x_bf, rot1_bf, nullptr, x_rot, DIMX, DIMX);
    // 2) xn = layernorm(x_rot)                    [2048, 768] bf16
    layernorm_to_bf16<<<BATCH, blk, 0, stream>>>(x_rot, ln_g, ln_b, xn_bf);
    // 3) h1 = relu(xn @ l1_w^T + l1_b)            [2048, 9216] bf16
    gemm_bf16_wmma<1, __bf16><<<dim3(MIDX / 128, BATCH / 128), blk, 0, stream>>>(
        xn_bf, l1w_bf, l1_b, h1_bf, MIDX, DIMX);
    // 4) mu = h1 @ mu_w^T + mu_b                  [2048, 6144] f32
    gemm_bf16_wmma<0, float><<<dim3(HX / 128, BATCH / 128), blk, 0, stream>>>(
        h1_bf, muw_bf, mu_b, mu_pre, HX, MIDX);
    // 5) sig_pre = h1 @ sig_w^T + sig_b           [2048, 6144] f32
    gemm_bf16_wmma<0, float><<<dim3(HX / 128, BATCH / 128), blk, 0, stream>>>(
        h1_bf, sigw_bf, sig_b, sig_pre, HX, MIDX);
    // 6) h = tanh(mu + exp(sig_pre) * noise)      bf16
    {
        int n = BATCH * HX;
        reparam_tanh_bf16<<<(n + 255) / 256, blk, 0, stream>>>(
            mu_pre, sig_pre, noise, h_bf, n);
    }
    // 7) h_rot = h @ rot2^T                       [2048, 6144] bf16
    gemm_bf16_wmma<0, __bf16><<<dim3(HX / 128, BATCH / 128), blk, 0, stream>>>(
        h_bf, rot2_bf, nullptr, hrot_bf, HX, HX);
    // 8) a = relu(h_rot @ l2_w^T + l2_b)          [2048, 12288] f32
    gemm_bf16_wmma<1, float><<<dim3(MKX / 128, BATCH / 128), blk, 0, stream>>>(
        hrot_bf, l2w_bf, l2_b, a_act, MKX, HX);
    // 9) PQ argmin over codes                     [2048, 48] int
    pq_argmin<<<(BATCH * PQ_M) / 8, blk, 0, stream>>>(x_rot, codebook, amin);
    // 10) hard gumbel-softmax -> d_out            [2048, 48, 256] f32
    gumbel_hard_out<<<(BATCH * PQ_M) / 8, blk, 0, stream>>>(
        a_act, gumbel, amin, interp, out);
}
